// ShardedLatentBottleneckModule_43078521978968
// MI455X (gfx1250) — compile-verified, matrix operand order checked by reading, not run on hardware
//
#include <hip/hip_runtime.h>
#include <hip/hip_bf16.h>

#define NUM_SHARDS   512
#define NUM_CLUSTERS 256
#define INPUT_SIZE   1024
#define OUTPUT_SIZE  1024

typedef __attribute__((ext_vector_type(16))) __bf16   v16bf;
typedef __attribute__((ext_vector_type(8)))  __bf16   v8bf;
typedef __attribute__((ext_vector_type(8)))  float    v8f;
typedef __attribute__((ext_vector_type(4)))  unsigned v4u;

// One block per sample. 512 threads = 16 wave32 waves.
// Phase 1: logit = key^T * W + bias   (W gathered by shard, [1024 x 256], c-fastest)
// Phase 2: lat_h = prob   * lat       (lat gathered by shard, [256 x 1024], o-fastest)
// WMMA 16x16x32 bf16: vector lives in A row 0 (bf16-staged in LDS, masked by a
// vector AND -> no exec divergence), parameter tile streams into B in its
// natural coalesced layout. D row 0 (VGPR0, lanes 0..15) is the output tile.
__global__ __launch_bounds__(512) void slb_kernel(
    const int*   __restrict__ shard,
    const float* __restrict__ key,
    const float* __restrict__ latent,
    const float* __restrict__ weight,
    const float* __restrict__ bias,
    float*       __restrict__ out_lat,
    float*       __restrict__ out_prob)
{
    __shared__ __bf16 s_keyb[INPUT_SIZE];     // key, pre-converted to bf16
    __shared__ __bf16 s_probb[NUM_CLUSTERS];  // prob, pre-converted to bf16
    __shared__ float  s_logit[NUM_CLUSTERS];
    __shared__ float  s_red[NUM_CLUSTERS];

    const int b    = blockIdx.x;
    const int tid  = threadIdx.x;
    const int lane = tid & 31;
    const int wave = tid >> 5;          // 0..15
    const int g    = (lane >> 4) & 1;   // half-wave group (ISA K interleave)
    const int nm   = lane & 15;         // N for B/D tiles; A row-0 mask uses nm==0
    const unsigned msk = (nm == 0) ? 0xFFFFFFFFu : 0u;  // keep row 0 only

    const int sh = shard[b];
    const float* Wp = weight + (size_t)sh * (INPUT_SIZE * NUM_CLUSTERS);
    const float* Lp = latent + (size_t)sh * (NUM_CLUSTERS * OUTPUT_SIZE);
    const float* Bp = bias   + (size_t)sh * NUM_CLUSTERS;

    for (int i = tid; i < INPUT_SIZE; i += 512)
        s_keyb[i] = (__bf16)key[(size_t)b * INPUT_SIZE + i];
    __syncthreads();

    // ---------------- Phase 1: logits, one 16-cluster tile per wave ----------
    {
        const int c0 = wave * 16;
        v8f acc = {};
        for (int i0 = 0; i0 < INPUT_SIZE; i0 += 32) {
            __builtin_prefetch(&Wp[(size_t)(i0 + 32) * NUM_CLUSTERS + c0 + nm], 0, 3);

            // A row 0 = key chunk: two aligned 16B LDS reads per lane, masked.
            v4u lo = __builtin_bit_cast(v4u, *(const v8bf*)&s_keyb[i0 + g * 8]);
            v4u hi = __builtin_bit_cast(v4u, *(const v8bf*)&s_keyb[i0 + 16 + g * 8]);
            lo &= msk;
            hi &= msk;
            const v8bf alo = __builtin_bit_cast(v8bf, lo);
            const v8bf ahi = __builtin_bit_cast(v8bf, hi);
            v16bf a, bm;
            const float* col = Wp + (size_t)i0 * NUM_CLUSTERS + c0 + nm;
#pragma unroll
            for (int j = 0; j < 8; ++j) {
                a[j]     = alo[j];
                a[j + 8] = ahi[j];
                bm[j]     = (__bf16)col[(size_t)(g * 8 + j)      * NUM_CLUSTERS];
                bm[j + 8] = (__bf16)col[(size_t)(16 + g * 8 + j) * NUM_CLUSTERS];
            }
            acc = __builtin_amdgcn_wmma_f32_16x16x32_bf16(false, a, false, bm,
                                                          (short)0, acc, false, false);
        }
        // D row M=0 lives in VGPR0 lanes 0..15 (N = lane)
        if (lane < 16)
            s_logit[c0 + lane] = acc[0] + Bp[c0 + lane];
    }
    __syncthreads();

    // ---------------- Softmax over 256 clusters ------------------------------
    if (tid < NUM_CLUSTERS) s_red[tid] = s_logit[tid];
    __syncthreads();
    for (int s = NUM_CLUSTERS / 2; s > 0; s >>= 1) {
        if (tid < s) s_red[tid] = fmaxf(s_red[tid], s_red[tid + s]);
        __syncthreads();
    }
    const float mx = s_red[0];
    __syncthreads();
    if (tid < NUM_CLUSTERS) {
        float e = __expf(s_logit[tid] - mx);
        s_logit[tid] = e;
        s_red[tid]   = e;
    }
    __syncthreads();
    for (int s = NUM_CLUSTERS / 2; s > 0; s >>= 1) {
        if (tid < s) s_red[tid] += s_red[tid + s];
        __syncthreads();
    }
    const float inv = 1.0f / s_red[0];
    __syncthreads();
    if (tid < NUM_CLUSTERS) {
        float p = s_logit[tid] * inv;
        s_probb[tid] = (__bf16)p;
        out_prob[(size_t)b * NUM_CLUSTERS + tid] = p;
    }
    __syncthreads();

    // ---------------- Phase 2: lat_h, four 16-output tiles per wave ----------
#pragma unroll
    for (int t = 0; t < 4; ++t) {
        const int o0 = (wave * 4 + t) * 16;
        v8f acc = {};
        for (int c0 = 0; c0 < NUM_CLUSTERS; c0 += 32) {
            __builtin_prefetch(&Lp[(size_t)(c0 + 32) * OUTPUT_SIZE + o0 + nm], 0, 3);

            v4u lo = __builtin_bit_cast(v4u, *(const v8bf*)&s_probb[c0 + g * 8]);
            v4u hi = __builtin_bit_cast(v4u, *(const v8bf*)&s_probb[c0 + 16 + g * 8]);
            lo &= msk;
            hi &= msk;
            const v8bf alo = __builtin_bit_cast(v8bf, lo);
            const v8bf ahi = __builtin_bit_cast(v8bf, hi);
            v16bf a, bm;
            const float* col = Lp + (size_t)c0 * OUTPUT_SIZE + o0 + nm;
#pragma unroll
            for (int j = 0; j < 8; ++j) {
                a[j]     = alo[j];
                a[j + 8] = ahi[j];
                bm[j]     = (__bf16)col[(size_t)(g * 8 + j)      * OUTPUT_SIZE];
                bm[j + 8] = (__bf16)col[(size_t)(16 + g * 8 + j) * OUTPUT_SIZE];
            }
            acc = __builtin_amdgcn_wmma_f32_16x16x32_bf16(false, a, false, bm,
                                                          (short)0, acc, false, false);
        }
        if (lane < 16)
            out_lat[(size_t)b * OUTPUT_SIZE + o0 + lane] = acc[0];
    }
}

extern "C" void kernel_launch(void* const* d_in, const int* in_sizes, int n_in,
                              void* d_out, int out_size, void* d_ws, size_t ws_size,
                              hipStream_t stream) {
    const int*   shard  = (const int*)  d_in[0];
    const float* key    = (const float*)d_in[1];
    const float* latent = (const float*)d_in[2];
    const float* weight = (const float*)d_in[3];
    const float* bias   = (const float*)d_in[4];

    const int batch = in_sizes[0];          // 1024
    float* out_lat  = (float*)d_out;        // [B, OUTPUT_SIZE]
    float* out_prob = out_lat + (size_t)batch * OUTPUT_SIZE;  // [B, NUM_CLUSTERS]

    slb_kernel<<<batch, 512, 0, stream>>>(shard, key, latent, weight, bias,
                                          out_lat, out_prob);
}
